// LinearAttention_10316511445033
// MI455X (gfx1250) — matrix operand order repaired and checked by reading, not compiled, not run on hardware
//
#include <hip/hip_runtime.h>

#define DIM   1024
#define Hh    16
#define HD    64
#define Bb    4
#define Nn    4096
#define BN    (Bb*Nn)      // 16384
#define BH    (Bb*Hh)      // 64
#define NSPLIT 8

typedef __attribute__((ext_vector_type(16))) __bf16 v16bf;
typedef __attribute__((ext_vector_type(8)))  __bf16 v8bf;
typedef __attribute__((ext_vector_type(8)))  float  v8f;

static __device__ __forceinline__ unsigned short f2b(float f) {
    unsigned int u = __float_as_uint(f);
    unsigned int r = u + 0x7FFFu + ((u >> 16) & 1u);
    return (unsigned short)(r >> 16);
}
static __device__ __forceinline__ float b2f(unsigned int bits) {
    return __uint_as_float(bits << 16);
}
static __device__ __forceinline__ v16bf cat16(v8bf lo, v8bf hi) {
    v16bf r;
#pragma unroll
    for (int i = 0; i < 8; ++i) { r[i] = lo[i]; r[i + 8] = hi[i]; }
    return r;
}
static __device__ __forceinline__ v8f wmma_bf16(v16bf a, v16bf b, v8f c) {
    return __builtin_amdgcn_wmma_f32_16x16x32_bf16(false, a, false, b, (short)0, c, false, false);
}

// ---------------- Kernel 1: fp32 -> bf16 convert + fragment swizzle ----------
// xbs: A-fragment layout.  tile index = rt*32 + kks  (rt<1024 row tiles, kks<32)
//   xbs[(tile*2 + c)*256 + lane*8 + e] = x[rt*16 + (lane&15)]
//                                         [kks*32 + 8*(lane>>4) + 16*c + e]
// wbs: B-fragment layout.  tile index = ctg*32 + kks (ctg<192 col tiles)
//   wbs[(tile*2 + c)*256 + lane*8 + e] = W[ctg*16 + (lane&15)]
//                                         [kks*32 + 8*(lane>>4) + 16*c + e]
__global__ void k_convert(const float* __restrict__ x,
                          const float* __restrict__ Wq,
                          const float* __restrict__ Wk,
                          const float* __restrict__ Wv,
                          unsigned short* __restrict__ xbs,
                          unsigned short* __restrict__ wbs) {
    const long CX = (long)BN * DIM;
    const long CW = 3L * DIM * DIM;
    long stride = (long)gridDim.x * blockDim.x;
    for (long i = (long)blockIdx.x * blockDim.x + threadIdx.x; i < CX + CW; i += stride) {
        if (i < CX) {
            long blk = i >> 8;  int r = (int)(i & 255);
            int lane = r >> 3, e = r & 7;
            long tile = blk >> 1; int c = (int)(blk & 1);
            long rt = tile >> 5;  int kks = (int)(tile & 31);
            int row = (int)(rt * 16) + (lane & 15);
            int col = kks * 32 + 8 * (lane >> 4) + 16 * c + e;
            xbs[i] = f2b(x[(size_t)row * DIM + col]);
        } else {
            long w = i - CX;
            long blk = w >> 8;  int r = (int)(w & 255);
            int lane = r >> 3, e = r & 7;
            long tile = blk >> 1; int c = (int)(blk & 1);
            int ctg = (int)(tile >> 5); int kks = (int)(tile & 31);
            int wrow = ctg * 16 + (lane & 15);
            int col = kks * 32 + 8 * (lane >> 4) + 16 * c + e;
            int sel = wrow >> 10, rr = wrow & 1023;
            const float* W = (sel == 0) ? Wq : (sel == 1) ? Wk : Wv;
            wbs[w] = f2b(W[(size_t)rr * DIM + col]);
        }
    }
}

// ---------------- Kernel 2: QKV projection (bf16 WMMA, 32x64 per wave) -------
// grid (BN/32, 48), block 32. Coalesced fragment loads, fused bias/relu.
// k/v written transposed [b,h,d,n] via packed b128 stores (lane's 8 acc rows
// are 8 consecutive n at fixed d).
__global__ __launch_bounds__(32)
void k_proj(const unsigned short* __restrict__ xbs_,
            const unsigned short* __restrict__ wbs_,
            const float* __restrict__ bq,
            const float* __restrict__ bk,
            const float* __restrict__ bv,
            unsigned short* __restrict__ qb,
            unsigned short* __restrict__ kT,
            unsigned short* __restrict__ vT) {
    const __bf16* xbs = (const __bf16*)xbs_;
    const __bf16* wbs = (const __bf16*)wbs_;
    int lane = threadIdx.x;
    int lo16 = lane & 15, hb = lane >> 4;
    int m0 = blockIdx.x * 32;
    int g  = blockIdx.y;                    // 0..47 -> 64-col group over 3*DIM
    int rt0 = m0 >> 4;

    v8f acc[2][4] = {};
    const __bf16* aT0 = xbs + (size_t)rt0 * 32 * 512 + lane * 8;
    const __bf16* aT1 = aT0 + 32 * 512;
    const __bf16* bT  = wbs + (size_t)g * 4 * 32 * 512 + lane * 8;
    for (int kks = 0; kks < 32; ++kks) {
        size_t ko = (size_t)kks * 512;
        v16bf a0 = cat16(*(const v8bf*)(aT0 + ko), *(const v8bf*)(aT0 + ko + 256));
        v16bf a1 = cat16(*(const v8bf*)(aT1 + ko), *(const v8bf*)(aT1 + ko + 256));
        __builtin_prefetch(aT0 + ko + 4 * 512, 0, 3);
        __builtin_prefetch(aT1 + ko + 4 * 512, 0, 3);
#pragma unroll
        for (int ct = 0; ct < 4; ++ct) {
            const __bf16* bp = bT + (size_t)ct * 32 * 512 + ko;
            v16bf b = cat16(*(const v8bf*)bp, *(const v8bf*)(bp + 256));
            acc[0][ct] = wmma_bf16(a0, b, acc[0][ct]);
            acc[1][ct] = wmma_bf16(a1, b, acc[1][ct]);
        }
    }
    int s = g / 16;                          // 0=q,1=k,2=v
    const float* bias = (s == 0) ? bq : (s == 1) ? bk : bv;
    if (s == 0) {
        // q: row-major bf16, 32B coalesced runs per store
#pragma unroll
        for (int ct = 0; ct < 4; ++ct) {
            int c = (g % 16) * 64 + ct * 16 + lo16;
            float bias_v = bias[c];
#pragma unroll
            for (int at = 0; at < 2; ++at)
#pragma unroll
                for (int r = 0; r < 8; ++r) {
                    float v = fmaxf(acc[at][ct][r] + bias_v, 0.0f);
                    int rowr = m0 + at * 16 + r + 8 * hb;
                    qb[(size_t)rowr * DIM + c] = f2b(v);
                }
        }
    } else {
        // k/v: transposed [b,h,d,n]; pack 8 consecutive n into one b128 store
        unsigned short* dst = (s == 1) ? kT : vT;
        int bi = m0 >> 12;
        int h  = g & 15;
        int nb = m0 & (Nn - 1);
#pragma unroll
        for (int ct = 0; ct < 4; ++ct) {
            int d = ct * 16 + lo16;
            float bias_v = bias[(g & 15) * 64 + ct * 16 + lo16];
#pragma unroll
            for (int at = 0; at < 2; ++at) {
                uint4 pk;
                unsigned int w[4];
#pragma unroll
                for (int r2 = 0; r2 < 4; ++r2) {
                    float v0 = acc[at][ct][2 * r2] + bias_v;
                    float v1 = acc[at][ct][2 * r2 + 1] + bias_v;
                    if (s == 1) { v0 = fmaxf(v0, 0.0f); v1 = fmaxf(v1, 0.0f); }
                    w[r2] = (unsigned int)f2b(v0) | ((unsigned int)f2b(v1) << 16);
                }
                pk.x = w[0]; pk.y = w[1]; pk.z = w[2]; pk.w = w[3];
                size_t idx = (((size_t)(bi * Hh + h)) * HD + d) * Nn
                           + nb + at * 16 + 8 * hb;
                *(uint4*)(dst + idx) = pk;
            }
        }
    }
}

// ---------------- Kernel 3: kv partials = K^T V over N (WMMA) ----------------
// grid (BH, HD/16, NSPLIT), block 32. Each wave: 16 d-rows x 64 p-cols.
__global__ __launch_bounds__(32)
void k_kv(const unsigned short* __restrict__ kT_,
          const unsigned short* __restrict__ vT_,
          float* __restrict__ kvpart) {
    const __bf16* kT = (const __bf16*)kT_;
    const __bf16* vT = (const __bf16*)vT_;
    int lane = threadIdx.x;
    int lo16 = lane & 15, hb = lane >> 4;
    int bh = blockIdx.x, dt = blockIdx.y, sp = blockIdx.z;
    int d = dt * 16 + lo16;
    int nbase0 = sp * (Nn / NSPLIT);

    v8f acc[4] = {};
    const __bf16* aBase = kT + ((size_t)bh * HD + d) * Nn;
    for (int nn = 0; nn < Nn / NSPLIT; nn += 32) {
        int nb = nbase0 + nn;
        v16bf a = cat16(*(const v8bf*)(aBase + nb + 16 * hb),
                        *(const v8bf*)(aBase + nb + 16 * hb + 8));
#pragma unroll
        for (int pt = 0; pt < 4; ++pt) {
            int p = pt * 16 + lo16;
            const __bf16* bp = vT + ((size_t)bh * HD + p) * Nn + nb + 16 * hb;
            v16bf b = cat16(*(const v8bf*)bp, *(const v8bf*)(bp + 8));
            acc[pt] = wmma_bf16(a, b, acc[pt]);
        }
    }
#pragma unroll
    for (int pt = 0; pt < 4; ++pt)
#pragma unroll
        for (int r = 0; r < 8; ++r) {
            int dd = dt * 16 + r + 8 * hb;
            int pp = pt * 16 + lo16;
            kvpart[(((size_t)sp * BH + bh) * HD + dd) * HD + pp] = acc[pt][r];
        }
}

// ---------------- Kernel 3b: reduce partials, emit kv^T in bf16 ---------------
__global__ void k_kvred(const float* __restrict__ kvpart,
                        unsigned short* __restrict__ kvbT) {
    int i = blockIdx.x * 256 + threadIdx.x;
    if (i >= BH * HD * HD) return;
    int bh = i / (HD * HD);
    int rem = i % (HD * HD);
    int d = rem / HD, p = rem % HD;
    float s = 0.f;
#pragma unroll
    for (int sp = 0; sp < NSPLIT; ++sp)
        s += kvpart[(((size_t)sp * BH + bh) * HD + d) * HD + p];
    kvbT[((size_t)bh * HD + p) * HD + d] = f2b(s);   // transposed: [p][d]
}

// ---------------- Kernel 4a: k_sum, one wave per (bh,d) row ------------------
__global__ __launch_bounds__(256)
void k_ksum(const unsigned short* __restrict__ kT,
            float* __restrict__ ksum) {
    int lane = threadIdx.x & 31, wid = threadIdx.x >> 5;
    int i = blockIdx.x * 8 + wid;            // row index < BH*HD = 4096
    const uint4* rp = (const uint4*)(kT + (size_t)i * Nn);
    float s = 0.f;
    for (int j = lane; j < Nn / 8; j += 32) {
        uint4 w = rp[j];
        s += b2f(w.x & 0xFFFFu) + b2f(w.x >> 16)
           + b2f(w.y & 0xFFFFu) + b2f(w.y >> 16)
           + b2f(w.z & 0xFFFFu) + b2f(w.z >> 16)
           + b2f(w.w & 0xFFFFu) + b2f(w.w >> 16);
    }
#pragma unroll
    for (int o = 1; o < 32; o <<= 1) s += __shfl_xor(s, o, 32);
    if (lane == 0) ksum[i] = s;
}

// ---------------- Kernel 4b: denominators, one wave per token ----------------
__global__ __launch_bounds__(256)
void k_den(const unsigned short* __restrict__ qb,
           const float* __restrict__ ksum,
           float* __restrict__ den) {
    int lane = threadIdx.x & 31, wid = threadIdx.x >> 5;
    int t = blockIdx.x * 8 + wid;            // token < BN
    int b = t >> 12;
#pragma unroll
    for (int j = 0; j < 4; ++j) {
        int off = j * 256 + lane * 8;        // element offset within row
        int h = off >> 6;
        int dl = off & 63;
        const float* ks = ksum + ((size_t)(b * Hh + h)) * HD + dl;
        uint4 w = *(const uint4*)(qb + (size_t)t * DIM + off);
        float s = b2f(w.x & 0xFFFFu) * ks[0] + b2f(w.x >> 16) * ks[1]
                + b2f(w.y & 0xFFFFu) * ks[2] + b2f(w.y >> 16) * ks[3]
                + b2f(w.z & 0xFFFFu) * ks[4] + b2f(w.z >> 16) * ks[5]
                + b2f(w.w & 0xFFFFu) * ks[6] + b2f(w.w >> 16) * ks[7];
        s += __shfl_xor(s, 1, 32);
        s += __shfl_xor(s, 2, 32);
        s += __shfl_xor(s, 4, 32);
        if ((lane & 7) == 0) den[(size_t)t * Hh + h] = fmaxf(s, 1e-6f);
    }
}

// ---------------- Kernel 5: ctx = Q*kv, divide, residual ---------------------
// grid (BN/16, H), block 32.
__global__ __launch_bounds__(32)
void k_ctx(const unsigned short* __restrict__ qb_,
           const unsigned short* __restrict__ kvbT_,
           const float* __restrict__ den,
           const float* __restrict__ x,
           float* __restrict__ y) {
    const __bf16* qb = (const __bf16*)qb_;
    const __bf16* kvbT = (const __bf16*)kvbT_;
    int lane = threadIdx.x;
    int lo16 = lane & 15, hb = lane >> 4;
    int m0 = blockIdx.x * 16;
    int h = blockIdx.y;
    int bh = (m0 >> 12) * Hh + h;
    int row = m0 + lo16;

    v8f acc[4] = {};
    const __bf16* aBase = qb + (size_t)row * DIM + h * HD;
#pragma unroll
    for (int k0 = 0; k0 < HD; k0 += 32) {
        v16bf a = cat16(*(const v8bf*)(aBase + k0 + 16 * hb),
                        *(const v8bf*)(aBase + k0 + 16 * hb + 8));
#pragma unroll
        for (int pt = 0; pt < 4; ++pt) {
            int p = pt * 16 + lo16;
            const __bf16* bp = kvbT + ((size_t)bh * HD + p) * HD + k0 + 16 * hb;
            v16bf b = cat16(*(const v8bf*)bp, *(const v8bf*)(bp + 8));
            acc[pt] = wmma_bf16(a, b, acc[pt]);
        }
    }
    float inv[8];
#pragma unroll
    for (int r = 0; r < 8; ++r) {
        int rowr = m0 + r + 8 * hb;
        inv[r] = 1.0f / den[(size_t)rowr * Hh + h];
    }
#pragma unroll
    for (int pt = 0; pt < 4; ++pt)
#pragma unroll
        for (int r = 0; r < 8; ++r) {
            int rowr = m0 + r + 8 * hb;
            int c = h * HD + pt * 16 + lo16;
            size_t idx = (size_t)rowr * DIM + c;
            y[idx] = acc[pt][r] * inv[r] + x[idx];
        }
}

// ---------------- Kernel 6: LayerNorm in place -------------------------------
__global__ __launch_bounds__(256)
void k_ln(float* __restrict__ y,
          const float* __restrict__ gamma,
          const float* __restrict__ beta) {
    __shared__ float s1[256], s2[256];
    int t = threadIdx.x;
    size_t base = (size_t)blockIdx.x * DIM;
    float4 v = *(const float4*)(y + base + t * 4);
    float sum = v.x + v.y + v.z + v.w;
    float sq  = v.x * v.x + v.y * v.y + v.z * v.z + v.w * v.w;
    s1[t] = sum; s2[t] = sq;
    __syncthreads();
    for (int off = 128; off > 0; off >>= 1) {
        if (t < off) { s1[t] += s1[t + off]; s2[t] += s2[t + off]; }
        __syncthreads();
    }
    float mu  = s1[0] * (1.0f / DIM);
    float var = s2[0] * (1.0f / DIM) - mu * mu;
    float rstd = rsqrtf(var + 1e-5f);
    float4 g = *(const float4*)(gamma + t * 4);
    float4 bt = *(const float4*)(beta + t * 4);
    float4 o;
    o.x = (v.x - mu) * rstd * g.x + bt.x;
    o.y = (v.y - mu) * rstd * g.y + bt.y;
    o.z = (v.z - mu) * rstd * g.z + bt.z;
    o.w = (v.w - mu) * rstd * g.w + bt.w;
    *(float4*)(y + base + t * 4) = o;
}

extern "C" void kernel_launch(void* const* d_in, const int* in_sizes, int n_in,
                              void* d_out, int out_size, void* d_ws, size_t ws_size,
                              hipStream_t stream) {
    const float* x     = (const float*)d_in[0];
    const float* Wq    = (const float*)d_in[1];
    const float* bq    = (const float*)d_in[2];
    const float* Wk    = (const float*)d_in[3];
    const float* bk    = (const float*)d_in[4];
    const float* Wv    = (const float*)d_in[5];
    const float* bv    = (const float*)d_in[6];
    const float* gamma = (const float*)d_in[7];
    const float* beta  = (const float*)d_in[8];
    float* out = (float*)d_out;

    // workspace layout (bytes)
    char* ws = (char*)d_ws;
    size_t oXB = 0;                                   // BN*DIM bf16 (A-frag swizzled)
    size_t oWB = oXB + (size_t)BN * DIM * 2;          // 3*DIM*DIM bf16 (B-frag swizzled)
    size_t oQB = oWB + (size_t)3 * DIM * DIM * 2;     // BN*DIM bf16 (row major)
    size_t oKT = oQB + (size_t)BN * DIM * 2;          // BN*DIM bf16 (b,h,d,n)
    size_t oVT = oKT + (size_t)BN * DIM * 2;          // BN*DIM bf16 (b,h,p,n)
    size_t oKVP = oVT + (size_t)BN * DIM * 2;         // NSPLIT*BH*HD*HD f32
    size_t oKVT = oKVP + (size_t)NSPLIT * BH * HD * HD * 4; // BH*HD*HD bf16
    size_t oKS  = oKVT + (size_t)BH * HD * HD * 2;    // BH*HD f32
    size_t oDEN = oKS + (size_t)BH * HD * 4;          // BN*Hh f32
    size_t need = oDEN + (size_t)BN * Hh * 4;
    if (ws_size < need) return;

    unsigned short* xbs  = (unsigned short*)(ws + oXB);
    unsigned short* wbs  = (unsigned short*)(ws + oWB);
    unsigned short* qb   = (unsigned short*)(ws + oQB);
    unsigned short* kT   = (unsigned short*)(ws + oKT);
    unsigned short* vT   = (unsigned short*)(ws + oVT);
    float*          kvp  = (float*)(ws + oKVP);
    unsigned short* kvbT = (unsigned short*)(ws + oKVT);
    float*          ksum = (float*)(ws + oKS);
    float*          den  = (float*)(ws + oDEN);

    k_convert<<<4096, 256, 0, stream>>>(x, Wq, Wk, Wv, xbs, wbs);
    k_proj<<<dim3(BN / 32, 48), 32, 0, stream>>>(xbs, wbs, bq, bk, bv, qb, kT, vT);
    k_kv<<<dim3(BH, HD / 16, NSPLIT), 32, 0, stream>>>(kT, vT, kvp);
    k_kvred<<<(BH * HD * HD + 255) / 256, 256, 0, stream>>>(kvp, kvbT);
    k_ksum<<<512, 256, 0, stream>>>(kT, ksum);
    k_den<<<BN / 8, 256, 0, stream>>>(qb, ksum, den);
    k_ctx<<<dim3(BN / 16, Hh), 32, 0, stream>>>(qb, kvbT, den, x, out);
    k_ln<<<BN, 256, 0, stream>>>(out, gamma, beta);
}
